// Attention_fusion_59837484368571
// MI455X (gfx1250) — matrix operand order repaired and checked by reading, not compiled, not run on hardware
//
#include <hip/hip_runtime.h>
#include <hip/hip_bf16.h>

typedef _Float16 v16h __attribute__((ext_vector_type(16)));
typedef _Float16 v8h  __attribute__((ext_vector_type(8)));
typedef float    v8f  __attribute__((ext_vector_type(8)));

#define N_NODES   100000
#define D_FEAT    256
#define M_ATT     256
#define CHUNKS    25          // split-K chunks; 25 * 4000 == 100000 exactly
#define KC        4000        // K elements per chunk (125 WMMA steps of 32)
#define NTILE     64          // N columns per workgroup
#define LDS_PITCH 40          // f16 per transposed-B row (pad 32 -> 40 vs bank conflicts)
#define LDS_BUF   (NTILE * LDS_PITCH)   // one staging buffer (f16 elems)

// Minimal CDNA5 split work-group barrier: only drain our own LDS stores
// (dscnt) before signaling -- do NOT drain loadcnt like __syncthreads() does,
// so the pipelined global staging/A loads keep flying across the barrier.
__device__ __forceinline__ void wg_barrier_lds() {
  asm volatile("s_wait_dscnt 0x0\n\t"
               "s_barrier_signal -1\n\t"
               "s_barrier_wait -1" ::: "memory");
}

// ---------------------------------------------------------------------------
// Kernel 1: split-K WMMA GEMM partials.  P[view][chunk][i][j] = sum over the
// chunk's K-range of W[i][k] * x_view[k][j], f16 inputs / f32 accumulate.
// Block = 512 threads = 16 waves; wave w owns rows [16w,16w+16) x 64 cols.
// Double-buffered LDS staging: one barrier per 32-K step, staging load for
// step i+1 issued before the barrier of step i.
// ---------------------------------------------------------------------------
__global__ __launch_bounds__(512)
void gemm_partial_kernel(const float* __restrict__ x1,
                         const float* __restrict__ x2,
                         const float* __restrict__ x3,
                         const float* __restrict__ W,
                         float* __restrict__ P) {
  const int chunk = blockIdx.x;   // 0..24
  const int ntile = blockIdx.y;   // 0..3
  const int view  = blockIdx.z;   // 0..2
  const float* __restrict__ X = (view == 0) ? x1 : (view == 1) ? x2 : x3;

  const int tid   = threadIdx.x;
  const int wave  = tid >> 5;     // wave32
  const int lane  = tid & 31;
  const int lrow  = lane & 15;
  const int khalf = lane >> 4;    // A/B operand half-wave split

  const int n0   = ntile * NTILE;
  const int k0   = chunk * KC;
  const int mrow = wave * 16 + lrow;   // W row this lane feeds into the A operand

  __shared__ _Float16 ldsB[2 * LDS_BUF];   // 10 KB: double-buffered x chunk, f16, transposed

  v8f acc[4] = {};   // four 16x16 f32 tiles along N

  // staging decomposition: 512 threads * float4 = 2048 f32 = 32 K-rows x 64 cols
  const int skrow = tid >> 4;          // 0..31  (K row within chunk step)
  const int sj0   = (tid & 15) * 4;    // 0..60  (column group)
  const float* __restrict__ xstage = X + n0 + sj0;

  // prologue: stage the first 32-K slab
  float4 xb = *(const float4*)(xstage + (size_t)(k0 + skrow) * D_FEAT);

  int cur = 0;
  for (int kk = k0; kk < k0 + KC; kk += 32) {
    // ---- commit staged slab into buf[cur] (f16, transposed / column-major)
    _Float16* __restrict__ bcur = &ldsB[cur * LDS_BUF];
    bcur[(sj0 + 0) * LDS_PITCH + skrow] = (_Float16)xb.x;
    bcur[(sj0 + 1) * LDS_PITCH + skrow] = (_Float16)xb.y;
    bcur[(sj0 + 2) * LDS_PITCH + skrow] = (_Float16)xb.z;
    bcur[(sj0 + 3) * LDS_PITCH + skrow] = (_Float16)xb.w;

    // ---- branchless pipelined staging load for the NEXT step (last iteration
    //      harmlessly reloads the first slab instead of going out of bounds)
    const int kn = (kk + 32 < k0 + KC) ? (kk + 32) : k0;
    xb = *(const float4*)(xstage + (size_t)(kn + skrow) * D_FEAT);

    // ---- A operand direct from global: ISA 16-bit A 16x32 layout.
    // lane<16: K = kk+0..7 and kk+16..23 ; lane>=16: K = kk+8..15 and kk+24..31
    const float* wrow = W + (size_t)mrow * N_NODES + kk + khalf * 8;
    const float4 a0 = *(const float4*)(wrow + 0);
    const float4 a1 = *(const float4*)(wrow + 4);
    const float4 a2 = *(const float4*)(wrow + 16);
    const float4 a3 = *(const float4*)(wrow + 20);
    __builtin_prefetch(wrow + 64, 0, 1);   // pull next-next K slice of this W row

    v16h A;
    A[0] = (_Float16)a0.x; A[1] = (_Float16)a0.y; A[2]  = (_Float16)a0.z; A[3]  = (_Float16)a0.w;
    A[4] = (_Float16)a1.x; A[5] = (_Float16)a1.y; A[6]  = (_Float16)a1.z; A[7]  = (_Float16)a1.w;
    A[8] = (_Float16)a2.x; A[9] = (_Float16)a2.y; A[10] = (_Float16)a2.z; A[11] = (_Float16)a2.w;
    A[12] = (_Float16)a3.x; A[13] = (_Float16)a3.y; A[14] = (_Float16)a3.z; A[15] = (_Float16)a3.w;

    wg_barrier_lds();   // buf[cur] slab visible in LDS (dscnt-only drain)

    // ---- B operands: lane<16 holds col=lrow K 0..7,16..23, lane>=16 holds K 8..15,24..31.
    // Issue all 8 ds_load_b128 first; sched_barrier keeps the WMMAs from being
    // interleaved back between the loads, so DS returns pipeline under WMMA
    // with partial s_wait_dscnt instead of 0x0 lockstep.
    const int koff = khalf * 8;
    v16h Bf[4];
#pragma unroll
    for (int s = 0; s < 4; ++s) {
      const int jj = s * 16 + lrow;
      const v8h blo = *(const v8h*)&bcur[jj * LDS_PITCH + koff];
      const v8h bhi = *(const v8h*)&bcur[jj * LDS_PITCH + koff + 16];
#pragma unroll
      for (int e = 0; e < 8; ++e) { Bf[s][e] = blo[e]; Bf[s][e + 8] = bhi[e]; }
    }
    __builtin_amdgcn_sched_barrier(0);
#pragma unroll
    for (int s = 0; s < 4; ++s)
      acc[s] = __builtin_amdgcn_wmma_f32_16x16x32_f16(false, A, false, Bf[s],
                                                      (short)0, acc[s], false, false);
    cur ^= 1;
  }

  // ---- C/D layout: VGPR r, lanes 0-15 -> M=r, lanes 16-31 -> M=8+r, N=lane&15
  float* pout = P + ((size_t)view * CHUNKS + chunk) * (M_ATT * (size_t)D_FEAT);
  const int rbase = wave * 16 + khalf * 8;
#pragma unroll
  for (int s = 0; s < 4; ++s) {
    const int col = n0 + s * 16 + lrow;
#pragma unroll
    for (int r = 0; r < 8; ++r)
      pout[(size_t)(rbase + r) * D_FEAT + col] = acc[s][r];
  }
}

// ---------------------------------------------------------------------------
// Kernel 2: scores[v][j] = sum_i h[i] * tanh( sum_c P[v][c][i][j] )
// One block per (view, column); 256 threads = one i each; LDS tree reduce.
// ---------------------------------------------------------------------------
__global__ __launch_bounds__(256)
void scores_kernel(const float* __restrict__ P,
                   const float* __restrict__ hn,
                   float* __restrict__ scores) {
  const int v = blockIdx.x >> 8;          // 0..2
  const int j = blockIdx.x & 255;         // 0..255
  const int i = threadIdx.x;

  const float* p = P + ((size_t)v * CHUNKS * M_ATT + i) * D_FEAT + j;
  float s = 0.0f;
#pragma unroll 5
  for (int c = 0; c < CHUNKS; ++c)
    s += p[(size_t)c * M_ATT * D_FEAT];

  __shared__ float red[256];
  red[i] = tanhf(s) * hn[i];
  __syncthreads();
  for (int off = 128; off > 0; off >>= 1) {
    if (i < off) red[i] += red[i + off];
    __syncthreads();
  }
  if (i == 0) scores[v * D_FEAT + j] = red[0];
}

// ---------------------------------------------------------------------------
// Kernel 3: out = softmax_over_views(scores)[col]-weighted sum of x1,x2,x3.
// float4 per thread, fully coalesced stream (the dominant 410 MB).
// ---------------------------------------------------------------------------
__device__ __forceinline__ void softmax3(float a, float b, float c,
                                         float& w0, float& w1, float& w2) {
  const float m  = fmaxf(a, fmaxf(b, c));
  const float e0 = __expf(a - m), e1 = __expf(b - m), e2 = __expf(c - m);
  const float inv = 1.0f / (e0 + e1 + e2);
  w0 = e0 * inv; w1 = e1 * inv; w2 = e2 * inv;
}

__global__ __launch_bounds__(256)
void fuse_out_kernel(const float* __restrict__ x1,
                     const float* __restrict__ x2,
                     const float* __restrict__ x3,
                     const float* __restrict__ scores,
                     float* __restrict__ out) {
  const size_t idx = (size_t)blockIdx.x * 256 + threadIdx.x;  // float4 index
  const int    j   = (int)(idx & 63) * 4;
  const size_t off = (idx >> 6) * D_FEAT + j;

  const float4 s0 = *(const float4*)(scores + j);
  const float4 s1 = *(const float4*)(scores + D_FEAT + j);
  const float4 s2 = *(const float4*)(scores + 2 * D_FEAT + j);
  const float4 a  = *(const float4*)(x1 + off);
  const float4 b  = *(const float4*)(x2 + off);
  const float4 c  = *(const float4*)(x3 + off);

  float w0, w1, w2;
  float4 o;
  softmax3(s0.x, s1.x, s2.x, w0, w1, w2); o.x = w0 * a.x + w1 * b.x + w2 * c.x;
  softmax3(s0.y, s1.y, s2.y, w0, w1, w2); o.y = w0 * a.y + w1 * b.y + w2 * c.y;
  softmax3(s0.z, s1.z, s2.z, w0, w1, w2); o.z = w0 * a.z + w1 * b.z + w2 * c.z;
  softmax3(s0.w, s1.w, s2.w, w0, w1, w2); o.w = w0 * a.w + w1 * b.w + w2 * c.w;
  *(float4*)(out + off) = o;
}

// ---------------------------------------------------------------------------
extern "C" void kernel_launch(void* const* d_in, const int* in_sizes, int n_in,
                              void* d_out, int out_size, void* d_ws, size_t ws_size,
                              hipStream_t stream) {
  (void)in_sizes; (void)n_in; (void)out_size; (void)ws_size;
  const float* x1 = (const float*)d_in[0];
  const float* x2 = (const float*)d_in[1];
  const float* x3 = (const float*)d_in[2];
  const float* W  = (const float*)d_in[3];
  const float* hn = (const float*)d_in[4];
  float* out = (float*)d_out;

  float* P      = (float*)d_ws;                                   // [3][25][256][256] f32
  float* scores = P + (size_t)3 * CHUNKS * M_ATT * D_FEAT;        // [3][256] f32

  dim3 g1(CHUNKS, 4, 3);                       // 300 workgroups x 16 waves
  gemm_partial_kernel<<<g1, 512, 0, stream>>>(x1, x2, x3, W, P);

  scores_kernel<<<dim3(3 * D_FEAT), 256, 0, stream>>>(P, hn, scores);

  const int nvec4 = (N_NODES * D_FEAT) / 4;    // 6,400,000
  fuse_out_kernel<<<dim3(nvec4 / 256), 256, 0, stream>>>(x1, x2, x3, scores, out);
}
